// Decoder_73959336837319
// MI455X (gfx1250) — matrix operand order repaired and checked by reading, not compiled, not run on hardware
//
#include <hip/hip_runtime.h>
#include <hip/hip_bf16.h>

// ---------------- problem constants ----------------
#define BB    64
#define TEE   500
#define TDD   400
#define DDIM  256
#define FDIM  400
#define MAXP  512

typedef __bf16 v16bf __attribute__((ext_vector_type(16)));
typedef float  v8f   __attribute__((ext_vector_type(8)));

union FragAB { v16bf v; uint4 q[2]; unsigned short u[16]; };

__device__ __forceinline__ float bf2f(unsigned short h) {
    return __uint_as_float(((unsigned int)h) << 16);
}
__device__ __forceinline__ unsigned short f2bf(float f) {
    unsigned int u = __float_as_uint(f);
    u += 0x7FFFu + ((u >> 16) & 1u);
    return (unsigned short)(u >> 16);
}

// gfx1250 async global->LDS copy (16B per lane), tracked by ASYNCcnt.
// ldsoff = LDS byte offset (low 32 bits of generic pointer to __shared__).
__device__ __forceinline__ void async_g2l_b128(unsigned ldsoff, const void* gaddr) {
    asm volatile("global_load_async_to_lds_b128 %0, %1, off"
                 :: "v"(ldsoff), "v"(gaddr) : "memory");
}
__device__ __forceinline__ void async_wait0() {
    asm volatile("s_wait_asynccnt 0x0" ::: "memory");
}

// ---------------- elementwise / prep kernels ----------------
__global__ __launch_bounds__(256) void k_cvt_bf16(const float* __restrict__ s,
                                                  unsigned short* __restrict__ d, int n) {
    int i = blockIdx.x * 256 + threadIdx.x;
    if (i < n) d[i] = f2bf(s[i]);
}

// inputs (B,TD,FD) f32 -> (B,FD,TD) bf16
__global__ __launch_bounds__(256) void k_inputsT(const float* __restrict__ in,
                                                 unsigned short* __restrict__ out) {
    int i = blockIdx.x * 256 + threadIdx.x;
    if (i >= BB * FDIM * TDD) return;
    int b = i / (FDIM * TDD);
    int r = i - b * FDIM * TDD;
    int f = r / TDD, t = r - f * TDD;
    out[i] = f2bf(in[(size_t)b * TDD * FDIM + (size_t)t * FDIM + f]);
}

// keys_pe bf16 (B,TE,D) = keys + key_pos_table[text_positions]
__global__ __launch_bounds__(256) void k_keys_pe(const float* __restrict__ keys,
                                                 const float* __restrict__ kpt,
                                                 const int* __restrict__ tpos,
                                                 unsigned short* __restrict__ out) {
    int i = blockIdx.x * 256 + threadIdx.x;
    if (i >= BB * TEE * DDIM) return;
    int b = i / (TEE * DDIM);
    int r = i - b * TEE * DDIM;
    int s = r / DDIM, d = r - s * DDIM;
    int p = tpos[b * TEE + s];
    out[i] = f2bf(keys[i] + kpt[(size_t)p * DDIM + d]);
}

// valsT bf16 (B,D,512) = values^T, zero-padded past TE
__global__ __launch_bounds__(256) void k_valsT(const float* __restrict__ vals,
                                               unsigned short* __restrict__ out) {
    int i = blockIdx.x * 256 + threadIdx.x;
    if (i >= BB * DDIM * 512) return;
    int b = i / (DDIM * 512);
    int r = i - b * DDIM * 512;
    int d = r / 512, sp = r - d * 512;
    unsigned short v = 0;
    if (sp < TEE) v = f2bf(vals[(size_t)b * TEE * DDIM + (size_t)sp * DDIM + d]);
    out[i] = v;
}

// q (B,TD,D) = encOut^T + query_pos_table[frame_positions]; write f32 + bf16
__global__ __launch_bounds__(256) void k_prep_q(const unsigned short* __restrict__ enc,
                                                const float* __restrict__ qpt,
                                                const int* __restrict__ fpos,
                                                float* __restrict__ qf,
                                                unsigned short* __restrict__ qb) {
    int i = blockIdx.x * 256 + threadIdx.x;
    if (i >= BB * TDD * DDIM) return;
    int b = i / (TDD * DDIM);
    int r = i - b * TDD * DDIM;
    int t = r / DDIM, d = r - t * DDIM;
    int p = fpos[b * TDD + t];
    float v = bf2f(enc[(size_t)b * DDIM * TDD + (size_t)d * TDD + t]) + qpt[(size_t)p * DDIM + d];
    qf[i] = v;
    qb[i] = f2bf(v);
}

// highway gate (in-place on x): x = sig(h1)*h2 + (1-sig(h1))*x
__global__ __launch_bounds__(256) void k_gate(unsigned short* __restrict__ x,
                                              const unsigned short* __restrict__ h) {
    int i = blockIdx.x * 256 + threadIdx.x;
    if (i >= BB * DDIM * TDD) return;
    int b = i / (DDIM * TDD);
    int ct = i - b * DDIM * TDD;
    const unsigned short* hb = h + (size_t)b * 2 * DDIM * TDD;
    float h1 = bf2f(hb[ct]);
    float h2 = bf2f(hb[ct + DDIM * TDD]);
    float xo = bf2f(x[i]);
    float s = 1.0f / (1.0f + __expf(-h1));
    x[i] = f2bf(s * h2 + (1.0f - s) * xo);
}

// copy encoder output into second half of concat buffer (channels 256..511)
__global__ __launch_bounds__(256) void k_concatQ(const unsigned short* __restrict__ q,
                                                 unsigned short* __restrict__ cc) {
    int i = blockIdx.x * 256 + threadIdx.x;
    if (i >= BB * DDIM * TDD) return;
    int b = i / (DDIM * TDD);
    int ct = i - b * DDIM * TDD;
    cc[(size_t)b * 2 * DDIM * TDD + DDIM * TDD + ct] = q[i];
}

// softmax over 500 scores per row; in-place f32 (d_out attn) + bf16 padded copy
__global__ __launch_bounds__(256) void k_softmax(float* __restrict__ scores,
                                                 unsigned short* __restrict__ abf) {
    int row = blockIdx.x;
    float* s = scores + (size_t)row * TEE;
    unsigned short* o = abf + (size_t)row * 512;
    __shared__ float red[256];
    int tid = threadIdx.x;
    float mx = -1e30f;
    for (int i = tid; i < TEE; i += 256) mx = fmaxf(mx, s[i]);
    red[tid] = mx; __syncthreads();
    for (int st = 128; st > 0; st >>= 1) { if (tid < st) red[tid] = fmaxf(red[tid], red[tid + st]); __syncthreads(); }
    mx = red[0]; __syncthreads();
    float sum = 0.f;
    for (int i = tid; i < TEE; i += 256) { float e = __expf(s[i] - mx); s[i] = e; sum += e; }
    red[tid] = sum; __syncthreads();
    for (int st = 128; st > 0; st >>= 1) { if (tid < st) red[tid] += red[tid + st]; __syncthreads(); }
    float inv = 1.0f / red[0];
    for (int i = tid; i < 512; i += 256) {
        if (i < TEE) { float a = s[i] * inv; s[i] = a; o[i] = f2bf(a); }
        else o[i] = 0;
    }
}

// done = sigmoid(xpre . fc_w + fc_b), one wave per (b,t)
__global__ __launch_bounds__(32) void k_done(const unsigned short* __restrict__ xpre,
                                             const float* __restrict__ fcw,
                                             const float* __restrict__ fcb,
                                             float* __restrict__ out) {
    int row = blockIdx.x;
    const unsigned short* xr = xpre + (size_t)row * FDIM;
    float sum = 0.f;
    for (int i = threadIdx.x; i < FDIM; i += 32) sum += bf2f(xr[i]) * fcw[i];
    #pragma unroll
    for (int o = 16; o > 0; o >>= 1) sum += __shfl_xor(sum, o, 32);
    if (threadIdx.x == 0) out[row] = 1.0f / (1.0f + __expf(-(sum + fcb[0])));
}

// ---------------- WMMA workhorse 1: conv-layout GEMM ----------------
// Y[b,o,t] = act( sum_k W[o,k] * Xg[b,k,t] + bias[o] )
// Xg gathers k->(c,j) with causal dilated shift when KW==3.
// MODE 0: linear->bf16(B,O,T); 1: relu->bf16(B,O,T); 2: sigmoid->f32(B,T,O) + bf16 pre(B,T,O)
template<int MODE>
__global__ __launch_bounds__(256) void gemm_ct(
    const unsigned short* __restrict__ W, const float* __restrict__ bias,
    const unsigned short* __restrict__ X,
    unsigned short* __restrict__ Ybf, float* __restrict__ Yf32,
    int O, int Ck, int CX, int T, int KW, int dil)
{
    __shared__ __align__(16) unsigned short lA[128 * 32];
    __shared__ __align__(16) unsigned short lB[32 * 32];
    const int b    = blockIdx.z;
    const int t0   = blockIdx.x * 32;
    const int o0   = blockIdx.y * 128;
    const int tid  = threadIdx.x;
    const int lane = tid & 31;
    const int wv   = tid >> 5;
    const unsigned short* Xb = X + (size_t)b * CX * T;
    v8f acc0 = {}; v8f acc1 = {};
    const int nk = (Ck + 31) >> 5;
    for (int ks = 0; ks < nk; ++ks) {
        const int k0 = ks << 5;
        __syncthreads();
        // ---- load A tile (128 rows x 32 K): async DMA to LDS on the fast path ----
        for (int i = tid; i < 512; i += 256) {
            int row = i >> 2, ch = i & 3;
            int gk = k0 + ch * 8;
            int o  = o0 + row;
            unsigned short* dst = lA + row * 32 + ch * 8;
            if (o < O && gk + 8 <= Ck) {
                async_g2l_b128((unsigned)(size_t)dst, W + (size_t)o * Ck + gk);
            } else {
                unsigned short tmp[8] __attribute__((aligned(16)));
                #pragma unroll
                for (int e = 0; e < 8; ++e)
                    tmp[e] = (o < O && gk + e < Ck) ? W[(size_t)o * Ck + gk + e] : (unsigned short)0;
                *(uint4*)dst = *(const uint4*)tmp;
            }
        }
        // prefetch next weight K-slab (speculative; lowers to global_prefetch_b8)
        if (k0 + 32 < Ck && (o0 + (tid >> 1)) < O)
            __builtin_prefetch(W + (size_t)(o0 + (tid >> 1)) * Ck + k0 + 32, 0, 1);
        // ---- load B tile (32 t x 32 K), gather with causal shift ----
        for (int i = tid; i < 1024; i += 256) {
            int tt = i >> 5, k = i & 31;
            int gk = k0 + k;
            unsigned short v = 0;
            if (gk < Ck && (t0 + tt) < T) {
                int c, ts;
                if (KW == 3) { c = gk / 3; int j = gk - c * 3; ts = t0 + tt - (2 - j) * dil; }
                else         { c = gk;                         ts = t0 + tt; }
                if (ts >= 0 && ts < T) v = Xb[(size_t)c * T + ts];
            }
            lB[tt * 32 + k] = v;
        }
        async_wait0();          // drain this wave's async DMA before publishing tile
        __syncthreads();
        // ---- fragments + 2 WMMAs ----
        FragAB fa, fb0, fb1;
        {
            int m = lane & 15, h = lane >> 4;
            const unsigned short* Ar = lA + (wv * 16 + m) * 32;
            fa.q[0] = *(const uint4*)(Ar + 16 * h);
            fa.q[1] = *(const uint4*)(Ar + 16 * h + 8);
            int n = lane & 15;
            const unsigned short* Br0 = lB + n * 32;
            const unsigned short* Br1 = lB + (16 + n) * 32;
            fb0.q[0] = *(const uint4*)(Br0 + 16 * h);
            fb0.q[1] = *(const uint4*)(Br0 + 16 * h + 8);
            fb1.q[0] = *(const uint4*)(Br1 + 16 * h);
            fb1.q[1] = *(const uint4*)(Br1 + 16 * h + 8);
        }
        acc0 = __builtin_amdgcn_wmma_f32_16x16x32_bf16(false, fa.v, false, fb0.v, (short)0, acc0, false, false);
        acc1 = __builtin_amdgcn_wmma_f32_16x16x32_bf16(false, fa.v, false, fb1.v, (short)0, acc1, false, false);
    }
    // ---- epilogue ----
    const int nl = lane & 15, hh = lane >> 4;
    const int ta = t0 + nl, tb = t0 + 16 + nl;
    #pragma unroll
    for (int r = 0; r < 8; ++r) {
        int o = o0 + wv * 16 + r + 8 * hh;
        if (o >= O) continue;
        float bv = bias ? bias[o] : 0.0f;
        float v0 = acc0[r] + bv;
        float v1 = acc1[r] + bv;
        if (MODE == 1) { v0 = fmaxf(v0, 0.0f); v1 = fmaxf(v1, 0.0f); }
        if (MODE == 2) {
            if (ta < T) {
                size_t off = (size_t)b * T * O + (size_t)ta * O + o;
                Yf32[off] = 1.0f / (1.0f + __expf(-v0));
                Ybf[off]  = f2bf(v0);
            }
            if (tb < T) {
                size_t off = (size_t)b * T * O + (size_t)tb * O + o;
                Yf32[off] = 1.0f / (1.0f + __expf(-v1));
                Ybf[off]  = f2bf(v1);
            }
        } else {
            size_t rowoff = (size_t)b * O * T + (size_t)o * T;
            if (ta < T) Ybf[rowoff + ta] = f2bf(v0);
            if (tb < T) Ybf[rowoff + tb] = f2bf(v1);
        }
    }
}

// ---------------- WMMA workhorse 2: row-major A(M,K) x B(N,K)^T ----------------
// MODE 0: p = A.Wq^T + bq        -> bf16 (B,M,256)
// MODE 1: scores = p.kpe^T       -> f32 d_out attn (B,M,500)
// MODE 2: ctx = attn.valsT^T*sq  -> bf16 (B,M,256)
// MODE 3: R = (ctx.Wo^T+bo+q)*c  -> bf16 transposed into concat (B,512,T) ch 0..255
template<int MODE>
__global__ __launch_bounds__(256) void gemm_rm(
    const unsigned short* __restrict__ A, size_t strideA,
    const unsigned short* __restrict__ Bm, size_t strideB,
    const float* __restrict__ bias, const float* __restrict__ qres,
    unsigned short* __restrict__ Ybf, float* __restrict__ Yf32,
    int M, int N, int K)
{
    __shared__ __align__(16) unsigned short lA[128 * 32];
    __shared__ __align__(16) unsigned short lB[16 * 32];
    const int b    = blockIdx.z;
    const int n0   = blockIdx.x * 16;
    const int m0   = blockIdx.y * 128;
    const int tid  = threadIdx.x;
    const int lane = tid & 31;
    const int wv   = tid >> 5;
    const unsigned short* Ab = A  + (size_t)b * strideA;
    const unsigned short* Bb = Bm + (size_t)b * strideB;
    v8f acc = {};
    const int nk = K >> 5;  // K is 256 or 512: always multiple of 32
    for (int ks = 0; ks < nk; ++ks) {
        const int k0 = ks << 5;
        __syncthreads();
        for (int i = tid; i < 512; i += 256) {
            int row = i >> 2, ch = i & 3;
            int m = m0 + row;
            unsigned short* dst = lA + row * 32 + ch * 8;
            if (m < M) {
                async_g2l_b128((unsigned)(size_t)dst, Ab + (size_t)m * K + k0 + ch * 8);
            } else {
                uint4 z = {0u, 0u, 0u, 0u};
                *(uint4*)dst = z;
            }
        }
        for (int i = tid; i < 64; i += 256) {
            int nn = i >> 2, ch = i & 3;
            int n = n0 + nn;
            unsigned short* dst = lB + nn * 32 + ch * 8;
            if (n < N) {
                async_g2l_b128((unsigned)(size_t)dst, Bb + (size_t)n * K + k0 + ch * 8);
            } else {
                uint4 z = {0u, 0u, 0u, 0u};
                *(uint4*)dst = z;
            }
        }
        async_wait0();
        __syncthreads();
        FragAB fa, fb;
        {
            int m = lane & 15, h = lane >> 4;
            const unsigned short* Ar = lA + (wv * 16 + m) * 32;
            fa.q[0] = *(const uint4*)(Ar + 16 * h);
            fa.q[1] = *(const uint4*)(Ar + 16 * h + 8);
            const unsigned short* Br = lB + (lane & 15) * 32;
            fb.q[0] = *(const uint4*)(Br + 16 * h);
            fb.q[1] = *(const uint4*)(Br + 16 * h + 8);
        }
        acc = __builtin_amdgcn_wmma_f32_16x16x32_bf16(false, fa.v, false, fb.v, (short)0, acc, false, false);
    }
    const int nl = lane & 15, hh = lane >> 4;
    const int ng = n0 + nl;
    #pragma unroll
    for (int r = 0; r < 8; ++r) {
        int m = m0 + wv * 16 + r + 8 * hh;
        if (m >= M || ng >= N) continue;
        float v = acc[r];
        if (MODE == 0) {
            v += bias[ng];
            Ybf[(size_t)b * M * DDIM + (size_t)m * DDIM + ng] = f2bf(v);
        } else if (MODE == 1) {
            Yf32[(size_t)b * M * TEE + (size_t)m * TEE + ng] = v;
        } else if (MODE == 2) {
            v *= 22.360679774997898f;  // sqrt(TE)
            Ybf[(size_t)b * M * DDIM + (size_t)m * DDIM + ng] = f2bf(v);
        } else {  // MODE 3
            v = (v + bias[ng] + qres[(size_t)b * M * DDIM + (size_t)m * DDIM + ng]) * 0.70710678118654752f;
            Ybf[(size_t)b * 2 * DDIM * TDD + (size_t)ng * TDD + m] = f2bf(v);  // R^T into concat ch 0..255
        }
    }
}

// ---------------- host-side orchestration ----------------
extern "C" void kernel_launch(void* const* d_in, const int* in_sizes, int n_in,
                              void* d_out, int out_size, void* d_ws, size_t ws_size,
                              hipStream_t stream) {
    (void)in_sizes; (void)n_in; (void)out_size; (void)ws_size;
    // inputs
    const float* keys   = (const float*)d_in[0];
    const float* values = (const float*)d_in[1];
    const float* inputs = (const float*)d_in[2];
    const float* kpt    = (const float*)d_in[3];
    const float* qpt    = (const float*)d_in[4];
    const float* ae_w0  = (const float*)d_in[5];   const float* ae_b0 = (const float*)d_in[6];
    const float* ae_w1  = (const float*)d_in[7];   const float* ae_b1 = (const float*)d_in[8];
    const float* ae_w2  = (const float*)d_in[9];   const float* ae_b2 = (const float*)d_in[10];
    const float* ae_hww = (const float*)d_in[11];  const float* ae_hwb = (const float*)d_in[12];
    const float* wq     = (const float*)d_in[13];  const float* bq = (const float*)d_in[14];
    const float* wo     = (const float*)d_in[15];  const float* bo = (const float*)d_in[16];
    const float* ad_w0  = (const float*)d_in[17];  const float* ad_b0 = (const float*)d_in[18];
    const float* ad_hww = (const float*)d_in[19];  const float* ad_hwb = (const float*)d_in[20];
    const float* ad_w1  = (const float*)d_in[21];  const float* ad_b1 = (const float*)d_in[22];
    const float* ad_w2  = (const float*)d_in[23];  const float* ad_b2 = (const float*)d_in[24];
    const float* ad_w3  = (const float*)d_in[25];  const float* ad_b3 = (const float*)d_in[26];
    const float* ad_w4  = (const float*)d_in[27];  const float* ad_b4 = (const float*)d_in[28];
    const float* fc_w   = (const float*)d_in[29];  const float* fc_b = (const float*)d_in[30];
    const int* tpos     = (const int*)d_in[31];
    const int* fpos     = (const int*)d_in[32];

    float* out_outputs = (float*)d_out;                               // (B,TD,FD)
    float* out_attn    = (float*)d_out + (size_t)BB * TDD * FDIM;     // (1,B,TD,TE)
    float* out_done    = out_attn + (size_t)BB * TDD * TEE;           // (B,TD,1)

    // workspace carve-up
    char* base = (char*)d_ws;
    size_t cur = 0;
    auto alloc = [&](size_t bytes) -> char* {
        char* p = base + cur;
        cur = (cur + bytes + 255) & ~(size_t)255;
        return p;
    };
    auto allocH = [&](size_t n) { return (unsigned short*)alloc(n * 2); };
    unsigned short* w_ae0  = allocH(256 * 400);
    unsigned short* w_ae1  = allocH(256 * 256);
    unsigned short* w_ae2  = allocH(256 * 256);
    unsigned short* w_aehw = allocH((size_t)10 * 512 * 768);
    unsigned short* w_wq   = allocH(256 * 256);
    unsigned short* w_wo   = allocH(256 * 256);
    unsigned short* w_ad0  = allocH(256 * 512);
    unsigned short* w_adhw = allocH((size_t)6 * 512 * 768);
    unsigned short* w_ad1  = allocH(256 * 256);
    unsigned short* w_ad2  = allocH(256 * 256);
    unsigned short* w_ad3  = allocH(256 * 256);
    unsigned short* w_ad4  = allocH(400 * 256);
    unsigned short* xinT   = allocH((size_t)BB * FDIM * TDD);
    unsigned short* actA   = allocH((size_t)BB * DDIM * TDD);
    unsigned short* actB   = allocH((size_t)BB * DDIM * TDD);
    unsigned short* hbuf   = allocH((size_t)BB * 2 * DDIM * TDD);
    unsigned short* concat = allocH((size_t)BB * 2 * DDIM * TDD);
    unsigned short* kpe_bf = allocH((size_t)BB * TEE * DDIM);
    unsigned short* valsT  = allocH((size_t)BB * DDIM * 512);
    float*          q_f32  = (float*)alloc((size_t)BB * TDD * DDIM * 4);
    unsigned short* q_bf   = allocH((size_t)BB * TDD * DDIM);
    unsigned short* p_bf   = allocH((size_t)BB * TDD * DDIM);
    unsigned short* atn_bf = allocH((size_t)BB * TDD * 512);
    unsigned short* ctx_bf = allocH((size_t)BB * TDD * DDIM);
    unsigned short* xpre   = allocH((size_t)BB * TDD * FDIM);

    auto cvt = [&](const float* s, unsigned short* d, int n) {
        k_cvt_bf16<<<(n + 255) / 256, 256, 0, stream>>>(s, d, n);
    };
    cvt(ae_w0, w_ae0, 256 * 400);
    cvt(ae_w1, w_ae1, 256 * 256);
    cvt(ae_w2, w_ae2, 256 * 256);
    cvt(ae_hww, w_aehw, 10 * 512 * 768);
    cvt(wq, w_wq, 256 * 256);
    cvt(wo, w_wo, 256 * 256);
    cvt(ad_w0, w_ad0, 256 * 512);
    cvt(ad_hww, w_adhw, 6 * 512 * 768);
    cvt(ad_w1, w_ad1, 256 * 256);
    cvt(ad_w2, w_ad2, 256 * 256);
    cvt(ad_w3, w_ad3, 256 * 256);
    cvt(ad_w4, w_ad4, 400 * 256);

    int nIn = BB * FDIM * TDD;
    k_inputsT<<<(nIn + 255) / 256, 256, 0, stream>>>(inputs, xinT);
    int nKp = BB * TEE * DDIM;
    k_keys_pe<<<(nKp + 255) / 256, 256, 0, stream>>>(keys, kpt, tpos, kpe_bf);
    int nVt = BB * DDIM * 512;
    k_valsT<<<(nVt + 255) / 256, 256, 0, stream>>>(values, valsT);

    const dim3 blk(256);
    const int  tTiles = (TDD + 31) / 32;                 // 13
    const dim3 g256(tTiles, 2, BB);                      // O=256
    const dim3 g512(tTiles, 4, BB);                      // O=512
    const dim3 g400(tTiles, 4, BB);                      // O=400 (guarded)
    const int  nEl = BB * DDIM * TDD;
    const dim3 gEl((nEl + 255) / 256);

    // ---- audio encoder ----
    gemm_ct<1><<<g256, blk, 0, stream>>>(w_ae0, ae_b0, xinT, actA, nullptr, 256, 400, 400, TDD, 1, 1);
    gemm_ct<1><<<g256, blk, 0, stream>>>(w_ae1, ae_b1, actA, actB, nullptr, 256, 256, 256, TDD, 1, 1);
    gemm_ct<0><<<g256, blk, 0, stream>>>(w_ae2, ae_b2, actB, actA, nullptr, 256, 256, 256, TDD, 1, 1);
    const int enc_dil[10] = {1, 3, 9, 27, 1, 3, 9, 27, 3, 3};
    for (int i = 0; i < 10; ++i) {
        gemm_ct<0><<<g512, blk, 0, stream>>>(w_aehw + (size_t)i * 512 * 768, ae_hwb + (size_t)i * 512,
                                             actA, hbuf, nullptr, 512, 768, 256, TDD, 3, enc_dil[i]);
        k_gate<<<gEl, blk, 0, stream>>>(actA, hbuf);
    }

    // ---- attention ----
    k_prep_q<<<gEl, blk, 0, stream>>>(actA, qpt, fpos, q_f32, q_bf);
    const size_t sQ = (size_t)TDD * DDIM;
    const size_t sK = (size_t)TEE * DDIM;
    const size_t sV = (size_t)DDIM * 512;
    const size_t sAt = (size_t)TDD * 512;
    gemm_rm<0><<<dim3(16, 4, BB), blk, 0, stream>>>(q_bf, sQ, w_wq, 0, bq, nullptr,
                                                    p_bf, nullptr, TDD, DDIM, DDIM);
    gemm_rm<1><<<dim3(32, 4, BB), blk, 0, stream>>>(p_bf, sQ, kpe_bf, sK, nullptr, nullptr,
                                                    nullptr, out_attn, TDD, TEE, DDIM);
    k_softmax<<<BB * TDD, blk, 0, stream>>>(out_attn, atn_bf);
    gemm_rm<2><<<dim3(16, 4, BB), blk, 0, stream>>>(atn_bf, sAt, valsT, sV, nullptr, nullptr,
                                                    ctx_bf, nullptr, TDD, DDIM, 512);
    gemm_rm<3><<<dim3(16, 4, BB), blk, 0, stream>>>(ctx_bf, sQ, w_wo, 0, bo, q_f32,
                                                    concat, nullptr, TDD, DDIM, DDIM);
    k_concatQ<<<gEl, blk, 0, stream>>>(actA, concat);

    // ---- audio decoder ----
    gemm_ct<0><<<g256, blk, 0, stream>>>(w_ad0, ad_b0, concat, actA, nullptr, 256, 512, 512, TDD, 1, 1);
    const int dec_dil[6] = {1, 3, 9, 27, 1, 1};
    for (int i = 0; i < 6; ++i) {
        gemm_ct<0><<<g512, blk, 0, stream>>>(w_adhw + (size_t)i * 512 * 768, ad_hwb + (size_t)i * 512,
                                             actA, hbuf, nullptr, 512, 768, 256, TDD, 3, dec_dil[i]);
        k_gate<<<gEl, blk, 0, stream>>>(actA, hbuf);
    }
    gemm_ct<1><<<g256, blk, 0, stream>>>(w_ad1, ad_b1, actA, actB, nullptr, 256, 256, 256, TDD, 1, 1);
    gemm_ct<1><<<g256, blk, 0, stream>>>(w_ad2, ad_b2, actB, actA, nullptr, 256, 256, 256, TDD, 1, 1);
    gemm_ct<1><<<g256, blk, 0, stream>>>(w_ad3, ad_b3, actA, actB, nullptr, 256, 256, 256, TDD, 1, 1);
    gemm_ct<2><<<g400, blk, 0, stream>>>(w_ad4, ad_b4, actB, xpre, out_outputs, 400, 256, 256, TDD, 1, 1);

    // ---- done flag ----
    k_done<<<BB * TDD, dim3(32), 0, stream>>>(xpre, fc_w, fc_b, out_done);
}
